// MS_SSA_Conv_26465588478215
// MI455X (gfx1250) — matrix operand order repaired and checked by reading, not compiled, not run on hardware
//
#include <hip/hip_runtime.h>

typedef __attribute__((ext_vector_type(16))) _Float16 v16h;
typedef __attribute__((ext_vector_type(8)))  _Float16 v8h;
typedef __attribute__((ext_vector_type(8)))  float    v8f;

#define TT 4
#define BB 8192
#define DD 128
#define HH 8
#define DHD 1024
#define NROW (TT*BB)   // 32768 rows, packed r = b*4 + t

// ---------------- prep kernels ----------------

__global__ void cvt_f16_kernel(const float* __restrict__ src,
                               _Float16* __restrict__ dst, int n) {
    int i = blockIdx.x * blockDim.x + threadIdx.x;
    if (i < n) dst[i] = (_Float16)src[i];
}

// fold bias into BN: y = acc*scale + shift, scale = g*rsqrt(v+eps),
// shift = (b - m)*scale + beta
__global__ void bn_prep_kernel(const float* __restrict__ b, const float* __restrict__ g,
                               const float* __restrict__ bet, const float* __restrict__ m,
                               const float* __restrict__ v,
                               float* __restrict__ scale, float* __restrict__ shift, int n) {
    int i = blockIdx.x * blockDim.x + threadIdx.x;
    if (i < n) {
        float s = g[i] * rsqrtf(v[i] + 1e-5f);
        scale[i] = s;
        shift[i] = (b[i] - m[i]) * s + bet[i];
    }
}

// LIF over T on input x (T,B,D) -> binary spikes xs in (b,t)-row order, f16
__global__ void lif_x_kernel(const float* __restrict__ x, _Float16* __restrict__ xs) {
    int i = blockIdx.x * blockDim.x + threadIdx.x;   // over B*D
    if (i >= BB * DD) return;
    int d = i & (DD - 1);
    int b = i >> 7;
    float vm = 0.f;
#pragma unroll
    for (int t = 0; t < TT; ++t) {
        float xv = x[(t * BB + b) * DD + d];
        vm = 0.5f * (vm + xv);                // v += (x - v)/tau, tau=2
        float s = (vm >= 1.0f) ? 1.0f : 0.0f; // heaviside(v - 1)
        xs[(b * TT + t) * DD + d] = (_Float16)s;
        vm = (s > 0.f) ? 0.f : vm;            // hard reset
    }
}

// ---------------- fused QKV + LIF + head-sum + kv-LIF + gating ----------------
// One wave computes a 16(M)x16(N) tile for all three branch GEMMs (K=128).
// Rows r = b*4 + t  =>  the 8 accumulator slots of a lane hold two complete
// 4-timestep groups => LIF scan is in-register. Columns n = d*8 + h => the
// 8-head reduction is a 3-step shfl_xor within an aligned 8-lane group.
__global__ __launch_bounds__(256)
void qkv_fused_kernel(const _Float16* __restrict__ xs,
                      const _Float16* __restrict__ wq,
                      const _Float16* __restrict__ wk,
                      const _Float16* __restrict__ wv,
                      const float* __restrict__ scq, const float* __restrict__ shq,
                      const float* __restrict__ sck, const float* __restrict__ shk,
                      const float* __restrict__ scv, const float* __restrict__ shv,
                      _Float16* __restrict__ a16,
                      float* __restrict__ v_out) {
    int wave = blockIdx.x * (blockDim.x >> 5) + (threadIdx.x >> 5);
    int lane = threadIdx.x & 31;
    int nT = wave & 63;          // 1024/16 = 64 column tiles
    int mT = wave >> 6;          // 32768/16 = 2048 row tiles
    int tileN = nT * 16, tileM = mT * 16;

    int l15 = lane & 15;
    int hi  = lane >> 4;
    int n   = tileN + l15;       // this lane's output column

    const _Float16* arow = xs + (size_t)(tileM + l15) * DD;
    int koff = hi * 16;          // A: lanes 16-31 hold the +16 K chunk

    v8f cq = {}, ck = {}, cv = {};
#pragma unroll
    for (int kk = 0; kk < 4; ++kk) {
        int k0 = kk * 32;
        // A fragment (16x32 f16): elems 0..15 = K[k0+koff..k0+koff+15]
        v8h a0 = *(const v8h*)(arow + k0 + koff);
        v8h a1 = *(const v8h*)(arow + k0 + 8 + koff);
        v16h af = __builtin_shufflevector(a0, a1, 0,1,2,3,4,5,6,7,8,9,10,11,12,13,14,15);
        // B fragment (32x16 f16): lane holds col n, 16 consecutive K from k0+hi*16
        int kb = k0 + hi * 16;
        v16h bq = *(const v16h*)(wq + (size_t)n * DD + kb);
        v16h bk = *(const v16h*)(wk + (size_t)n * DD + kb);
        v16h bv = *(const v16h*)(wv + (size_t)n * DD + kb);
        cq = __builtin_amdgcn_wmma_f32_16x16x32_f16(false, af, false, bq, (short)0, cq, false, false);
        ck = __builtin_amdgcn_wmma_f32_16x16x32_f16(false, af, false, bk, (short)0, ck, false, false);
        cv = __builtin_amdgcn_wmma_f32_16x16x32_f16(false, af, false, bv, (short)0, cv, false, false);
    }

    float sq_ = scq[n], hq_ = shq[n];
    float sk_ = sck[n], hk_ = shk[n];
    float sv_ = scv[n], hv_ = shv[n];

    float sq[8], sk[8], sv[8];
#pragma unroll
    for (int c = 0; c < 2; ++c) {
        float vq = 0.f, vk = 0.f, vv = 0.f;
#pragma unroll
        for (int t = 0; t < 4; ++t) {
            int j = c * 4 + t;
            float y, s;
            y = cq[j] * sq_ + hq_; vq = 0.5f * (vq + y);
            s = (vq >= 1.f) ? 1.f : 0.f; sq[j] = s; vq = (s > 0.f) ? 0.f : vq;
            y = ck[j] * sk_ + hk_; vk = 0.5f * (vk + y);
            s = (vk >= 1.f) ? 1.f : 0.f; sk[j] = s; vk = (s > 0.f) ? 0.f : vk;
            y = cv[j] * sv_ + hv_; vv = 0.5f * (vv + y);
            s = (vv >= 1.f) ? 1.f : 0.f; sv[j] = s; vv = (s > 0.f) ? 0.f : vv;
        }
    }

    int rbase = tileM + 8 * hi;     // C layout: VGPR j = row rbase + j
    int hh = n & 7, dcol = n >> 3;  // n = d*8 + h

    // v spikes are output #2: (T,B,H,D) fp32
#pragma unroll
    for (int j = 0; j < 8; ++j) {
        int r = rbase + j, b = r >> 2, t = r & 3;
        v_out[((size_t)(t * BB + b) * HH + hh) * DD + dcol] = sv[j];
    }

    // kv[t,b,d] = sum over 8 heads of k*v : reduce within aligned 8-lane group
    float kvp[8];
#pragma unroll
    for (int j = 0; j < 8; ++j) {
        float p = sk[j] * sv[j];
        p += __shfl_xor(p, 1);
        p += __shfl_xor(p, 2);
        p += __shfl_xor(p, 4);
        kvp[j] = p;
    }
    // talking-heads LIF, v_th = 0.5
    float akv[8];
#pragma unroll
    for (int c = 0; c < 2; ++c) {
        float vm = 0.f;
#pragma unroll
        for (int t = 0; t < 4; ++t) {
            int j = c * 4 + t;
            vm = 0.5f * (vm + kvp[j]);
            float s = (vm >= 0.5f) ? 1.f : 0.f;
            akv[j] = s;
            vm = (s > 0.f) ? 0.f : vm;
        }
    }

    // a[r, h*128+d] = q * kv  (binary => exact in f16)
    int pcol = hh * DD + dcol;
#pragma unroll
    for (int j = 0; j < 8; ++j) {
        int r = rbase + j;
        a16[(size_t)r * DHD + pcol] = (_Float16)(sq[j] * akv[j]);
    }
}

// ---------------- projection GEMM + BN + residual ----------------
// out[t,b,:] = bn(a @ Wp.T) + x ; M=32768, K=1024, N=128
__global__ __launch_bounds__(256)
void proj_kernel(const _Float16* __restrict__ a16, const _Float16* __restrict__ wp,
                 const float* __restrict__ scp, const float* __restrict__ shp,
                 const float* __restrict__ x, float* __restrict__ out) {
    int wave = blockIdx.x * (blockDim.x >> 5) + (threadIdx.x >> 5);
    int lane = threadIdx.x & 31;
    int nT = wave & 7;           // 128/16 = 8 column tiles
    int mT = wave >> 3;          // 2048 row tiles
    int tileN = nT * 16, tileM = mT * 16;

    int l15 = lane & 15;
    int hi  = lane >> 4;
    int n   = tileN + l15;

    const _Float16* arow = a16 + (size_t)(tileM + l15) * DHD;
    int koff = hi * 16;

    v8f acc = {};
#pragma unroll 4
    for (int kk = 0; kk < 32; ++kk) {
        int k0 = kk * 32;
        v8h a0 = *(const v8h*)(arow + k0 + koff);
        v8h a1 = *(const v8h*)(arow + k0 + 8 + koff);
        v16h af = __builtin_shufflevector(a0, a1, 0,1,2,3,4,5,6,7,8,9,10,11,12,13,14,15);
        v16h bf = *(const v16h*)(wp + (size_t)n * DHD + k0 + hi * 16);
        acc = __builtin_amdgcn_wmma_f32_16x16x32_f16(false, af, false, bf, (short)0, acc, false, false);
    }

    float sc = scp[n], sh = shp[n];
    int rbase = tileM + 8 * hi;
#pragma unroll
    for (int j = 0; j < 8; ++j) {
        int r = rbase + j, b = r >> 2, t = r & 3;
        size_t idx = (size_t)(t * BB + b) * DD + n;
        out[idx] = acc[j] * sc + sh + x[idx];
    }
}

// ---------------- host launch ----------------

extern "C" void kernel_launch(void* const* d_in, const int* in_sizes, int n_in,
                              void* d_out, int out_size, void* d_ws, size_t ws_size,
                              hipStream_t stream) {
    const float* x   = (const float*)d_in[0];
    const float* Wq  = (const float*)d_in[1];
    const float* Wk  = (const float*)d_in[2];
    const float* Wv  = (const float*)d_in[3];
    const float* Wp  = (const float*)d_in[4];
    const float* bq  = (const float*)d_in[5],  *gq = (const float*)d_in[6];
    const float* betq= (const float*)d_in[7],  *mq = (const float*)d_in[8],  *vq = (const float*)d_in[9];
    const float* bk  = (const float*)d_in[10], *gk = (const float*)d_in[11];
    const float* betk= (const float*)d_in[12], *mk = (const float*)d_in[13], *vk = (const float*)d_in[14];
    const float* bv  = (const float*)d_in[15], *gv = (const float*)d_in[16];
    const float* betv= (const float*)d_in[17], *mv = (const float*)d_in[18], *vv = (const float*)d_in[19];
    const float* bp  = (const float*)d_in[20], *gp = (const float*)d_in[21];
    const float* betp= (const float*)d_in[22], *mp = (const float*)d_in[23], *vp = (const float*)d_in[24];

    // workspace layout
    char* ws = (char*)d_ws;
    _Float16* xs16 = (_Float16*)ws;                              //  8,388,608 B
    _Float16* a16  = (_Float16*)(ws + (size_t)8388608);          // 67,108,864 B
    char* wbase    = ws + (size_t)75497472;
    _Float16* wq16 = (_Float16*)(wbase);                         // 262,144 B each
    _Float16* wk16 = wq16 + 131072;
    _Float16* wv16 = wk16 + 131072;
    _Float16* wp16 = wv16 + 131072;
    float* prm = (float*)(wbase + (size_t)4 * 262144);
    float *scq = prm,        *shq = prm + 1024;
    float *sck = prm + 2048, *shk = prm + 3072;
    float *scv = prm + 4096, *shv = prm + 5120;
    float *scp = prm + 6144, *shp = prm + 6272;

    float* out_p = (float*)d_out;                 // (T,B,D)
    float* v_out = out_p + (size_t)TT * BB * DD;  // (T,B,H,D)

    // weight conversion fp32 -> f16
    cvt_f16_kernel<<<512, 256, 0, stream>>>(Wq, wq16, 131072);
    cvt_f16_kernel<<<512, 256, 0, stream>>>(Wk, wk16, 131072);
    cvt_f16_kernel<<<512, 256, 0, stream>>>(Wv, wv16, 131072);
    cvt_f16_kernel<<<512, 256, 0, stream>>>(Wp, wp16, 131072);

    // BN folding
    bn_prep_kernel<<<4, 256, 0, stream>>>(bq, gq, betq, mq, vq, scq, shq, 1024);
    bn_prep_kernel<<<4, 256, 0, stream>>>(bk, gk, betk, mk, vk, sck, shk, 1024);
    bn_prep_kernel<<<4, 256, 0, stream>>>(bv, gv, betv, mv, vv, scv, shv, 1024);
    bn_prep_kernel<<<1, 256, 0, stream>>>(bp, gp, betp, mp, vp, scp, shp, 128);

    // shortcut LIF -> xs spikes (f16, (b,t)-row order)
    lif_x_kernel<<<(BB * DD) / 256, 256, 0, stream>>>(x, xs16);

    // fused QKV: 2048 x 64 tiles, 8 waves/block
    qkv_fused_kernel<<<16384, 256, 0, stream>>>(xs16, wq16, wk16, wv16,
                                                scq, shq, sck, shk, scv, shv,
                                                a16, v_out);

    // projection: 2048 x 8 tiles, 8 waves/block
    proj_kernel<<<2048, 256, 0, stream>>>(a16, wp16, scp, shp, x, out_p);
}
